// DA_RNN_19387482374828
// MI455X (gfx1250) — matrix operand order repaired and checked by reading, not compiled, not run on hardware
//
#include <hip/hip_runtime.h>
#include <hip/hip_bf16.h>
#include <stdint.h>

// ---------------------------------------------------------------------------
// DA-RNN (B=512, T-1=31, D=256, HE=HD=512) for gfx1250 (MI455X).
// bf16 WMMA (v_wmma_f32_16x16x32_bf16) for all GEMMs, f32 VALU for LSTM /
// softmax epilogues. Decoder attention matmul algebraically split so the
// enc_out part is precomputed once (saves ~750 GFLOP vs reference).
// ---------------------------------------------------------------------------

typedef __bf16 bf16;
typedef __attribute__((ext_vector_type(16))) __bf16 v16bf;
typedef __attribute__((ext_vector_type(8)))  __bf16 v8bf;
typedef __attribute__((ext_vector_type(8)))  float  v8f;

__device__ __forceinline__ bf16 f2bf(float x) {
  __hip_bfloat16 h = __float2bfloat16(x);
  bf16 r; __builtin_memcpy(&r, &h, sizeof(r)); return r;
}
__device__ __forceinline__ float bf2f(bf16 x) {
  __hip_bfloat16 h; __builtin_memcpy(&h, &x, sizeof(h)); return __bfloat162float(h);
}
__device__ __forceinline__ float sigm(float x) { return 1.0f / (1.0f + __expf(-x)); }

// butterfly reductions inside a 16-lane half (wave32: xor masks < 16 stay in group)
__device__ __forceinline__ float red16(float v) {
  v += __shfl_xor(v, 8); v += __shfl_xor(v, 4);
  v += __shfl_xor(v, 2); v += __shfl_xor(v, 1); return v;
}
__device__ __forceinline__ float red16max(float v) {
  v = fmaxf(v, __shfl_xor(v, 8)); v = fmaxf(v, __shfl_xor(v, 4));
  v = fmaxf(v, __shfl_xor(v, 2)); v = fmaxf(v, __shfl_xor(v, 1)); return v;
}

// A fragment (16x32, row-major source): ISA 7.12.2 16-bit A layout.
// lanes 0-15 hold K {0..7,16..23}; lanes 16-31 hold K {8..15,24..31}.
__device__ __forceinline__ v16bf load_frag_a(const bf16* src, int ld, int lane) {
  int r = lane & 15, g = lane >> 4;
  const bf16* p = src + (size_t)r * ld + g * 8;
  v8bf lo = *(const v8bf*)p;
  v8bf hi = *(const v8bf*)(p + 16);
  return __builtin_shufflevector(lo, hi, 0,1,2,3,4,5,6,7,8,9,10,11,12,13,14,15);
}
// B fragment (32x16). B[k][n] = W[n][k] (weights row-major, row = output col).
// lanes 0-15 hold K {0..7,16..23}; lanes 16-31 hold K {8..15,24..31} (matches A).
__device__ __forceinline__ v16bf load_frag_b(const bf16* src, int ld, int lane) {
  int n = lane & 15, g = lane >> 4;
  const v8bf* p = (const v8bf*)(src + (size_t)n * ld + g * 8);
  v8bf lo = p[0], hi = p[2];
  return __builtin_shufflevector(lo, hi, 0,1,2,3,4,5,6,7,8,9,10,11,12,13,14,15);
}
#define WMMA_BF16(A, B, C) \
  __builtin_amdgcn_wmma_f32_16x16x32_bf16(false, (A), false, (B), (short)0, (C), false, false)

// ---------------------------------------------------------------------------
__global__ __launch_bounds__(256) void k_zero(uint32_t* p, size_t n) {
  size_t i = (size_t)blockIdx.x * 256 + threadIdx.x;
  if (i < n) p[i] = 0u;
}

// convert f32 weights -> bf16 workspace copies (encoder [Wih|Whh] fused)
__global__ __launch_bounds__(256) void k_cvt(
    const float* __restrict__ eWih, const float* __restrict__ eWhh,
    const float* __restrict__ dWhh, const float* __restrict__ dW1,
    bf16* __restrict__ Wcat, bf16* __restrict__ dWhhB,
    bf16* __restrict__ W1hs, bf16* __restrict__ W1enc) {
  const int N1 = 2048 * 768;
  const int N2 = N1 + 2048 * 512;
  const int N3 = N2 + 512 * 1024;
  const int N4 = N3 + 512 * 512;
  int i = blockIdx.x * 256 + threadIdx.x;
  if (i < N1) {
    int n = i / 768, k = i % 768;
    float v = (k < 256) ? eWih[n * 256 + k] : eWhh[n * 512 + (k - 256)];
    Wcat[i] = f2bf(v);
  } else if (i < N2) {
    int j = i - N1; dWhhB[j] = f2bf(dWhh[j]);
  } else if (i < N3) {
    int j = i - N2; int e = j / 1024, k = j % 1024;
    W1hs[j] = f2bf(dW1[e * 1536 + k]);
  } else if (i < N4) {
    int j = i - N3; int e = j / 512, k = j % 512;
    W1enc[j] = f2bf(dW1[e * 1536 + 1024 + k]);
  }
}

// t2[b,d,s] = sum_t X[b,t,d] * Wd[s,t] + bd[s]
__global__ __launch_bounds__(256) void k_t2(
    const float* __restrict__ inputs, const float* __restrict__ Wd,
    const float* __restrict__ bd, float* __restrict__ t2) {
  __shared__ float sWd[31 * 31];
  __shared__ float sBd[31];
  int tid = threadIdx.x, b = blockIdx.x;
  for (int i = tid; i < 31 * 31; i += 256) sWd[i] = Wd[i];
  if (tid < 31) sBd[tid] = bd[tid];
  __syncthreads();
  int d = tid;
  float acc[31];
#pragma unroll
  for (int s = 0; s < 31; ++s) acc[s] = sBd[s];
  for (int tt = 0; tt < 31; ++tt) {
    float x = inputs[((size_t)b * 31 + tt) * 257 + 1 + d];
#pragma unroll
    for (int s = 0; s < 31; ++s) acc[s] += x * sWd[s * 31 + tt];
  }
  float* o = t2 + ((size_t)b * 256 + d) * 31;
#pragma unroll
  for (int s = 0; s < 31; ++s) o[s] = acc[s];
}

// encoder input attention: writes w = softmax_d(score) * x_t as bf16 into A buffer cols 0..255
__global__ __launch_bounds__(256) void k_enc_attn(
    const float* __restrict__ inputs, const float* __restrict__ eh, const float* __restrict__ ec,
    const float* __restrict__ Wc, const float* __restrict__ bc, const float* __restrict__ t2,
    const float* __restrict__ Wa, const float* __restrict__ ba,
    bf16* __restrict__ Aw, int t) {
  __shared__ float sT1[16][31];
  __shared__ float sWa[31];
  int tid = threadIdx.x;
  int m0 = blockIdx.x * 16;
  if (tid < 31) sWa[tid] = Wa[tid];
  int r = tid >> 4, q = tid & 15;
  int b = m0 + r;
  const float* hrow = eh + (size_t)b * 512;
  const float* crow = ec + (size_t)b * 512;
  for (int s = 0; s < 31; ++s) {                 // t1 = [h,c] @ Wc.T + bc
    const float* wc = Wc + (size_t)s * 1024;
    float p = 0.f;
#pragma unroll 8
    for (int i = q * 32; i < q * 32 + 32; ++i) p += hrow[i] * wc[i] + crow[i] * wc[512 + i];
    p = red16(p);
    if (q == 0) sT1[r][s] = p + bc[s];
  }
  __syncthreads();
  float sc[16];
  float mx = -3.0e38f;
  float ba0 = ba[0];
#pragma unroll
  for (int i = 0; i < 16; ++i) {                 // score over d = q*16 + i
    int d = q * 16 + i;
    const float* t2p = t2 + ((size_t)b * 256 + d) * 31;
    float a = ba0;
#pragma unroll
    for (int s = 0; s < 31; ++s) a += tanhf(sT1[r][s] + t2p[s]) * sWa[s];
    sc[i] = a; mx = fmaxf(mx, a);
  }
  mx = red16max(mx);
  float sum = 0.f;
#pragma unroll
  for (int i = 0; i < 16; ++i) { sc[i] = __expf(sc[i] - mx); sum += sc[i]; }
  sum = red16(sum);
  float inv = 1.f / sum;
#pragma unroll
  for (int i = 0; i < 16; ++i) {
    int d = q * 16 + i;
    float xt = inputs[((size_t)b * 31 + t) * 257 + 1 + d];
    Aw[(size_t)b * 768 + d] = f2bf(sc[i] * inv * xt);
  }
}

// Fused gate GEMM + LSTM cell. Block = 16 batch rows x 64 cols of each gate group.
// gates = A(bf16, KD) @ W(bf16, 2048xKD).T [+ r1x*r1w rank-1] + bih + bhh
template <int KD>
__global__ __launch_bounds__(256) void k_lstm_step(
    const bf16* __restrict__ A, int lda,
    const bf16* __restrict__ W,
    const float* __restrict__ bih, const float* __restrict__ bhh,
    const float* __restrict__ r1x, const float* __restrict__ r1w,
    float* __restrict__ cbuf, float* __restrict__ hbuf,
    bf16* __restrict__ ob1, int ldo1,
    bf16* __restrict__ ob2, int ldo2, int ob2_is_c) {
  __shared__ bf16 sA[16 * KD];
  __shared__ float sG[4][16][64];
  int tid = threadIdx.x, wave = tid >> 5, lane = tid & 31;
  int m0 = blockIdx.x * 16;
  int n0 = blockIdx.y * 64;
  const int KV = KD / 8;
  for (int idx = tid; idx < 16 * KV; idx += 256) {   // stage A tile (b128 copies)
    int r = idx / KV, c = idx % KV;
    *(uint4*)&sA[r * KD + c * 8] = *(const uint4*)(A + (size_t)(m0 + r) * lda + c * 8);
  }
  __syncthreads();
  int nt0 = wave * 2, nt1 = wave * 2 + 1;            // 16 N-tiles: group = nt>>2, sub = nt&3
  int g0 = nt0 >> 2, s0 = nt0 & 3, g1 = nt1 >> 2, s1 = nt1 & 3;
  const bf16* W0 = W + (size_t)(g0 * 512 + n0 + s0 * 16) * KD;
  const bf16* W1 = W + (size_t)(g1 * 512 + n0 + s1 * 16) * KD;
  v8f acc0 = {}, acc1 = {};
  for (int k = 0; k < KD; k += 32) {
    __builtin_prefetch(W0 + k + 256, 0, 1);
    v16bf a  = load_frag_a(sA + k, KD, lane);
    v16bf b0 = load_frag_b(W0 + k, KD, lane);
    v16bf b1 = load_frag_b(W1 + k, KD, lane);
    acc0 = WMMA_BF16(a, b0, acc0);
    acc1 = WMMA_BF16(a, b1, acc1);
  }
  {
    int n = lane & 15, g = lane >> 4;
#pragma unroll
    for (int v = 0; v < 8; ++v) {
      int rr = v + 8 * g;
      sG[g0][rr][s0 * 16 + n] = acc0[v];
      sG[g1][rr][s1 * 16 + n] = acc1[v];
    }
  }
  __syncthreads();
  for (int idx = tid; idx < 1024; idx += 256) {      // fused LSTM epilogue
    int r = idx >> 6, j = idx & 63;
    int b = m0 + r;
    int col = n0 + j;
    float gi = sG[0][r][j] + bih[col]        + bhh[col];
    float gf = sG[1][r][j] + bih[512 + col]  + bhh[512 + col];
    float gg = sG[2][r][j] + bih[1024 + col] + bhh[1024 + col];
    float go = sG[3][r][j] + bih[1536 + col] + bhh[1536 + col];
    if (r1x) {
      float x = r1x[b];
      gi += x * r1w[col];        gf += x * r1w[512 + col];
      gg += x * r1w[1024 + col]; go += x * r1w[1536 + col];
    }
    size_t ix = (size_t)b * 512 + col;
    float c2 = sigm(gf) * cbuf[ix] + sigm(gi) * tanhf(gg);
    float h2 = sigm(go) * tanhf(c2);
    cbuf[ix] = c2; hbuf[ix] = h2;
    ob1[(size_t)b * ldo1 + col] = f2bf(h2);
    ob2[(size_t)b * ldo2 + col] = f2bf(ob2_is_c ? c2 : h2);
  }
}

// pre_enc = enc_out @ W1enc.T + b1   (M = B*31, N = 512, K = 512)
__global__ __launch_bounds__(256) void k_preenc(
    const bf16* __restrict__ Eh, const bf16* __restrict__ W,
    const float* __restrict__ b1, float* __restrict__ P) {
  __shared__ bf16 sA[16 * 512];
  int tid = threadIdx.x, wave = tid >> 5, lane = tid & 31;
  size_t m0 = (size_t)blockIdx.x * 16;
  int e0 = blockIdx.y * 128 + wave * 16;
  for (int idx = tid; idx < 16 * 64; idx += 256) {
    int r = idx / 64, c = idx % 64;
    *(uint4*)&sA[r * 512 + c * 8] = *(const uint4*)(Eh + (m0 + r) * 512 + c * 8);
  }
  __syncthreads();
  v8f acc = {};
  for (int k = 0; k < 512; k += 32) {
    v16bf a  = load_frag_a(sA + k, 512, lane);
    v16bf bb = load_frag_b(W + (size_t)e0 * 512 + k, 512, lane);
    acc = WMMA_BF16(a, bb, acc);
  }
  int n = lane & 15, g = lane >> 4;
#pragma unroll
  for (int v = 0; v < 8; ++v) {
    size_t row = m0 + v + 8 * g;
    P[row * 512 + e0 + n] = acc[v] + b1[e0 + n];
  }
}

// Fused decoder attention: hsW = [h,c] @ W1hs.T (WMMA) -> score/softmax/ctx/y_tilde
__global__ __launch_bounds__(256) void k_dec_attn(
    const bf16* __restrict__ dhc, const bf16* __restrict__ W1hs,
    const float* __restrict__ preenc, const float* __restrict__ W2,
    const float* __restrict__ b2, const bf16* __restrict__ ehist,
    const float* __restrict__ inputs, const float* __restrict__ fcW,
    const float* __restrict__ fcb, float* __restrict__ ctx,
    float* __restrict__ ytil, int t) {
  __shared__ bf16 sA[16 * 1024];
  __shared__ float sH[16][512];
  __shared__ float sS[16][31];
  int tid = threadIdx.x, wave = tid >> 5, lane = tid & 31;
  int m0 = blockIdx.x * 16;
  for (int idx = tid; idx < 16 * 128; idx += 256) {
    int r = idx >> 7, c = idx & 127;
    *(uint4*)&sA[r * 1024 + c * 8] = *(const uint4*)(dhc + (size_t)(m0 + r) * 1024 + c * 8);
  }
  __syncthreads();
  v8f acc[4] = {};
  for (int k = 0; k < 1024; k += 32) {
    v16bf a = load_frag_a(sA + k, 1024, lane);
#pragma unroll
    for (int j = 0; j < 4; ++j) {
      int e0 = (wave * 4 + j) * 16;
      v16bf bb = load_frag_b(W1hs + (size_t)e0 * 1024 + k, 1024, lane);
      acc[j] = WMMA_BF16(a, bb, acc[j]);
    }
  }
  {
    int n = lane & 15, g = lane >> 4;
#pragma unroll
    for (int j = 0; j < 4; ++j) {
      int e = (wave * 4 + j) * 16 + n;
#pragma unroll
      for (int v = 0; v < 8; ++v) sH[v + 8 * g][e] = acc[j][v];  // no +b1 (in preenc)
    }
  }
  __syncthreads();
  int r = tid >> 4, q = tid & 15;
  int b = m0 + r;
  for (int tt = 0; tt < 31; ++tt) {              // score[b,tt]
    const float* pe = preenc + ((size_t)b * 31 + tt) * 512;
    float p = 0.f;
#pragma unroll 8
    for (int i = q * 32; i < q * 32 + 32; ++i) p += tanhf(pe[i] + sH[r][i]) * W2[i];
    p = red16(p);
    if (q == 0) sS[r][tt] = p + b2[0];
  }
  __syncthreads();
  float mx = -3.0e38f;
  for (int tt = 0; tt < 31; ++tt) mx = fmaxf(mx, sS[r][tt]);
  float a31[31]; float sum = 0.f;
  for (int tt = 0; tt < 31; ++tt) { a31[tt] = __expf(sS[r][tt] - mx); sum += a31[tt]; }
  float inv = 1.0f / sum;
  float cx[32];
#pragma unroll
  for (int i = 0; i < 32; ++i) cx[i] = 0.f;
  for (int tt = 0; tt < 31; ++tt) {              // ctx = sum_t a * enc_out
    float aw = a31[tt] * inv;
    const bf16* ehp = ehist + ((size_t)b * 31 + tt) * 512 + q * 32;
#pragma unroll 8
    for (int i = 0; i < 32; ++i) cx[i] += aw * bf2f(ehp[i]);
  }
  float yp = 0.f;
#pragma unroll 8
  for (int i = 0; i < 32; ++i) {
    ctx[(size_t)b * 512 + q * 32 + i] = cx[i];
    yp += cx[i] * fcW[q * 32 + i];
  }
  yp = red16(yp);
  if (q == 0) {
    float yh = inputs[((size_t)b * 31 + t) * 257];  // y_hist[b, t]
    ytil[b] = yp + yh * fcW[512] + fcb[0];
  }
}

__global__ __launch_bounds__(128) void k_final(
    const float* __restrict__ dh, const float* __restrict__ ctx,
    const float* __restrict__ fcfW, const float* __restrict__ fcfb,
    float* __restrict__ out) {
  __shared__ float red[4];
  int b = blockIdx.x, tid = threadIdx.x;
  float p = 0.f;
  for (int i = tid; i < 512; i += 128)
    p += dh[(size_t)b * 512 + i] * fcfW[i] + ctx[(size_t)b * 512 + i] * fcfW[512 + i];
  p += __shfl_xor(p, 16); p += __shfl_xor(p, 8); p += __shfl_xor(p, 4);
  p += __shfl_xor(p, 2);  p += __shfl_xor(p, 1);
  if ((tid & 31) == 0) red[tid >> 5] = p;
  __syncthreads();
  if (tid == 0) out[b] = red[0] + red[1] + red[2] + red[3] + fcfb[0];
}

// ---------------------------------------------------------------------------
extern "C" void kernel_launch(void* const* d_in, const int* in_sizes, int n_in,
                              void* d_out, int out_size, void* d_ws, size_t ws_size,
                              hipStream_t stream) {
  (void)in_sizes; (void)n_in; (void)out_size;
  const float* inputs = (const float*)d_in[0];
  const float* eWih = (const float*)d_in[1];
  const float* eWhh = (const float*)d_in[2];
  const float* ebih = (const float*)d_in[3];
  const float* ebhh = (const float*)d_in[4];
  const float* eWc  = (const float*)d_in[5];
  const float* ebc  = (const float*)d_in[6];
  const float* eWd  = (const float*)d_in[7];
  const float* ebd  = (const float*)d_in[8];
  const float* eWa  = (const float*)d_in[9];
  const float* eba  = (const float*)d_in[10];
  const float* dW1  = (const float*)d_in[11];
  const float* db1  = (const float*)d_in[12];
  const float* dW2  = (const float*)d_in[13];
  const float* db2  = (const float*)d_in[14];
  const float* dWih = (const float*)d_in[15];
  const float* dWhh = (const float*)d_in[16];
  const float* dbih = (const float*)d_in[17];
  const float* dbhh = (const float*)d_in[18];
  const float* fcW  = (const float*)d_in[19];
  const float* fcb  = (const float*)d_in[20];
  const float* fcfW = (const float*)d_in[21];
  const float* fcfb = (const float*)d_in[22];

  char* ws = (char*)d_ws;
  size_t off = 0;
  auto take = [&](size_t bytes) -> char* {
    char* p = ws + off;
    off += (bytes + 255) & ~(size_t)255;
    return p;
  };
  bf16*  Wcat   = (bf16*)take(2048ull * 768 * 2);
  bf16*  dWhhB  = (bf16*)take(2048ull * 512 * 2);
  bf16*  W1hs   = (bf16*)take(512ull * 1024 * 2);
  bf16*  W1enc  = (bf16*)take(512ull * 512 * 2);
  float* t2     = (float*)take(512ull * 256 * 31 * 4);
  char*  zbase  = ws + off;                 // ---- zero-init region start ----
  bf16*  Abuf0  = (bf16*)take(512ull * 768 * 2);
  bf16*  Abuf1  = (bf16*)take(512ull * 768 * 2);
  float* eh     = (float*)take(512ull * 512 * 4);
  float* ec     = (float*)take(512ull * 512 * 4);
  float* dh     = (float*)take(512ull * 512 * 4);
  float* dc     = (float*)take(512ull * 512 * 4);
  bf16*  dhc0   = (bf16*)take(512ull * 1024 * 2);
  bf16*  dhc1   = (bf16*)take(512ull * 1024 * 2);
  size_t zbytes = (size_t)((ws + off) - zbase);  // ---- zero region end ----
  bf16*  ehist  = (bf16*)take(512ull * 31 * 512 * 2);
  float* preenc = (float*)take(512ull * 31 * 512 * 4);
  float* ctx    = (float*)take(512ull * 512 * 4);
  float* ytil   = (float*)take(512ull * 4);
  if (off > ws_size) return;  // insufficient scratch

  bf16* Abuf[2] = {Abuf0, Abuf1};
  bf16* dhc[2]  = {dhc0, dhc1};

  size_t zwords = zbytes / 4;
  k_zero<<<(unsigned)((zwords + 255) / 256), 256, 0, stream>>>((uint32_t*)zbase, zwords);

  const size_t NCVT = 2048ull * 768 + 2048ull * 512 + 512ull * 1024 + 512ull * 512;
  k_cvt<<<(unsigned)((NCVT + 255) / 256), 256, 0, stream>>>(
      eWih, eWhh, dWhh, dW1, Wcat, dWhhB, W1hs, W1enc);

  k_t2<<<512, 256, 0, stream>>>(inputs, eWd, ebd, t2);

  // -------- encoder: 31 sequential steps --------
  for (int t = 0; t < 31; ++t) {
    k_enc_attn<<<32, 256, 0, stream>>>(inputs, eh, ec, eWc, ebc, t2, eWa, eba,
                                       Abuf[t & 1], t);
    k_lstm_step<768><<<dim3(32, 8), 256, 0, stream>>>(
        Abuf[t & 1], 768, Wcat, ebih, ebhh,
        nullptr, nullptr, ec, eh,
        Abuf[(t + 1) & 1] + 256, 768,           // next step's A h-part (bf16)
        ehist + t * 512, 31 * 512, 0);          // enc_out history (bf16)
  }

  // -------- precompute enc_out @ W1enc.T + b1 (once) --------
  k_preenc<<<dim3(992, 4), 256, 0, stream>>>(ehist, W1enc, db1, preenc);

  // -------- decoder: 31 sequential steps --------
  for (int t = 0; t < 31; ++t) {
    k_dec_attn<<<32, 256, 0, stream>>>(dhc[t & 1], W1hs, preenc, dW2, db2,
                                       ehist, inputs, fcW, fcb, ctx, ytil, t);
    k_lstm_step<512><<<dim3(32, 8), 256, 0, stream>>>(
        dhc[t & 1], 1024, dWhhB, dbih, dbhh,
        ytil, dWih, dc, dh,
        dhc[(t + 1) & 1], 1024,                 // next [h|..] bf16
        dhc[(t + 1) & 1] + 512, 1024, 1);       // next [..|c] bf16
  }

  k_final<<<512, 128, 0, stream>>>(dh, ctx, fcfW, fcfb, (float*)d_out);
}
